// Backbone_4243427688698
// MI455X (gfx1250) — matrix operand order repaired and checked by reading, not compiled, not run on hardware
//
#include <hip/hip_runtime.h>
#include <hip/hip_bf16.h>
#include <math.h>

typedef __attribute__((ext_vector_type(16))) __bf16 v16bf;
typedef __attribute__((ext_vector_type(8)))  float  v8f;
typedef __attribute__((ext_vector_type(4)))  float  v4f;

#define D_MODEL 128
#define HID     512
#define NPG     9
#define EPG     16
#define GPW     16                  // graphs per workgroup
#define NODES   (GPW*NPG)           // 144 rows = 9 tiles of 16
#define NWAVES  9
#define NTHREADS (NWAVES*32)        // 288
#define STR     136                 // bf16 elems per padded row (272 B, bank-conflict free)
#define LN_EPS  1e-5f
#define NLAYERS 4

// workspace element offsets (bf16 elements)
#define WG_OFF 0
#define W1_OFF (NLAYERS*D_MODEL*D_MODEL)        // 65536
#define W2_OFF (W1_OFF + NLAYERS*D_MODEL*HID)   // 327680

// dynamic LDS byte offsets (all 16B aligned)
#define OFF_H   0                    // 144*136*2 = 39168   h (bf16)
#define OFF_A   39168                // 39168               agg (bf16, WMMA A source)
#define OFF_SCR 78336                // 73728               union: fp32 agg | deg/isd | FFN t | readout
#define OFF_EN  152064               // 1024                edge norms
#define OFF_SW  153088               // 640                 self-loop weights
#define OFF_SRC 153728               // 1024                local src
#define OFF_DST 154752               // 1024                local dst
#define SMEM_BYTES 155776

// ---------------------------------------------------------------------------
// Re-layout fp32 weight [Ktot][Ntot] into bf16 WMMA-B fragment blocks.
// Block = (kt, nf): 32 lanes x 16 bf16 (32 B/lane, contiguous per lane).
// Lane L holds column N = nf*16 + (L&15); VGPR j packs K = base+2j+o, base+2j+1+o (o = j<4 ? 0 : 8)
// with base = (L<16 ? 0 : 8). blockIdx.y = layer.
// ---------------------------------------------------------------------------
__global__ void prep_weights(const float* __restrict__ W, __bf16* __restrict__ out,
                             int Ktot, int Ntot) {
    const int total = Ktot * Ntot;
    const int tid = blockIdx.x * blockDim.x + threadIdx.x;
    if (tid >= total) return;
    const float* Wl = W + (size_t)blockIdx.y * total;
    __bf16*      ol = out + (size_t)blockIdx.y * total;
    int blk  = tid >> 9;           // 512 elems per 1024 B block
    int rem  = tid & 511;
    int lane = rem >> 4;
    int e    = rem & 15;
    int j = e >> 1, ph = e & 1;
    int nnf = Ntot >> 4;
    int kt = blk / nnf, nf = blk % nnf;
    int k = kt*32 + ((lane < 16) ? 0 : 8) + ((2*j + ph < 8) ? (2*j + ph) : (2*j + ph + 8));
    int n = nf*16 + (lane & 15);
    ol[tid] = (__bf16)Wl[k * Ntot + n];
}

// A fragment (16x32 bf16): lane<16: K 0..7 | 16..23 ; lane>=16: K 8..15 | 24..31
__device__ __forceinline__ v16bf load_frag_a(const __bf16* rowPtr, int kt, int lhalf) {
    union { v16bf v; v4f f[2]; } u;
    const __bf16* p = rowPtr + kt*32 + (lhalf ? 8 : 0);
    u.f[0] = *(const v4f*)p;
    u.f[1] = *(const v4f*)(p + 16);
    return u.v;
}
// B fragment: 32 B per lane, contiguous, pre-swizzled by prep_weights
__device__ __forceinline__ v16bf load_frag_b(const __bf16* base, int blk, int lane) {
    union { v16bf v; v4f f[2]; } u;
    const v4f* p = (const v4f*)(base + blk*512 + lane*16);
    u.f[0] = p[0];
    u.f[1] = p[1];
    return u.v;
}

__global__ __launch_bounds__(NTHREADS)
void gcn_fused_kernel(const float* __restrict__ op_table,
                      const float* __restrict__ device_emb,
                      const float* __restrict__ bg,
                      const float* __restrict__ ln_g,
                      const float* __restrict__ ln_b,
                      const float* __restrict__ b1,
                      const float* __restrict__ b2,
                      const float* __restrict__ fc_w,
                      const float* __restrict__ fc_b,
                      const int*   __restrict__ op_idx,
                      const int*   __restrict__ srcE,
                      const int*   __restrict__ dstE,
                      const __bf16* __restrict__ wsW,
                      float* __restrict__ out) {
    extern __shared__ v4f smemv[];
    char* smem = (char*)smemv;
    __bf16* hB    = (__bf16*)(smem + OFF_H);
    __bf16* aB    = (__bf16*)(smem + OFF_A);
    float*  scr   = (float*)(smem + OFF_SCR);
    float*  enorm = (float*)(smem + OFF_EN);
    float*  selfw = (float*)(smem + OFF_SW);
    int*    lsrc  = (int*)(smem + OFF_SRC);
    int*    ldst  = (int*)(smem + OFF_DST);

    const int tid = threadIdx.x;
    const int g0 = blockIdx.x * GPW;
    const int nodeBase = g0 * NPG;

    // ---- degrees / edge norms (scr used as scratch: deg at [0], isd at [256]) ----
    float* degF = scr;
    float* isd  = scr + 256;
    if (tid < NODES) degF[tid] = 1.0f;            // self loop
    __syncthreads();
    if (tid < GPW*EPG) {
        int e = g0*EPG + tid;
        int ls = srcE[e] - nodeBase;
        int ld = dstE[e] - nodeBase;
        lsrc[tid] = ls; ldst[tid] = ld;
        atomicAdd(&degF[ld], 1.0f);
    }
    __syncthreads();
    if (tid < NODES) isd[tid] = rsqrtf(degF[tid]);
    __syncthreads();
    if (tid < NODES) selfw[tid] = isd[tid] * isd[tid];
    if (tid < GPW*EPG) enorm[tid] = isd[lsrc[tid]] * isd[ldst[tid]];

    // ---- embedding: h = op_table[op] + device_emb ----
    for (int i = tid; i < NODES*D_MODEL; i += NTHREADS) {
        int n = i >> 7, c = i & 127;
        int op = op_idx[nodeBase + n];
        hB[n*STR + c] = (__bf16)(op_table[op*D_MODEL + c] + device_emb[c]);
    }
    __syncthreads();

    const int wave  = tid >> 5;
    const int lane  = tid & 31;
    const int lhalf = (lane >= 16) ? 1 : 0;
    const int l16   = lane & 15;
    const int row0  = wave * 16;

    #pragma unroll 1
    for (int l = 0; l < NLAYERS; ++l) {
        const __bf16* Wgl = wsW + WG_OFF + l*(D_MODEL*D_MODEL);
        const __bf16* W1l = wsW + W1_OFF + l*(D_MODEL*HID);
        const __bf16* W2l = wsW + W2_OFF + l*(HID*D_MODEL);
        __builtin_prefetch(Wgl, 0, 1);
        __builtin_prefetch(W1l, 0, 1);
        __builtin_prefetch(W2l, 0, 1);

        // ---- message passing: aggF = selfw*h + scatter(enorm*h[src]) ----
        float* aggF = scr;
        for (int i = tid; i < NODES*D_MODEL; i += NTHREADS) {
            int n = i >> 7, c = i & 127;
            aggF[i] = selfw[n] * (float)hB[n*STR + c];
        }
        __syncthreads();
        for (int i = tid; i < GPW*EPG*32; i += NTHREADS) {
            int e  = i >> 5;
            int cg = (i & 31) << 2;
            float w = enorm[e];
            const __bf16* hs = hB + lsrc[e]*STR + cg;
            float* ad = aggF + ldst[e]*D_MODEL + cg;
            atomicAdd(ad+0, w*(float)hs[0]);
            atomicAdd(ad+1, w*(float)hs[1]);
            atomicAdd(ad+2, w*(float)hs[2]);
            atomicAdd(ad+3, w*(float)hs[3]);
        }
        __syncthreads();
        for (int i = tid; i < NODES*D_MODEL; i += NTHREADS) {
            int n = i >> 7, c = i & 127;
            aB[n*STR + c] = (__bf16)aggF[i];
        }
        __syncthreads();

        // ---- GEMM: relu(agg @ Wg + bg) ---- (wave owns rows row0..row0+15)
        const __bf16* aRow = aB + (row0 + l16)*STR;
        v16bf aF[4];
        #pragma unroll
        for (int kt = 0; kt < 4; ++kt) aF[kt] = load_frag_a(aRow, kt, lhalf);

        v8f c[8];
        #pragma unroll
        for (int nf = 0; nf < 8; ++nf) {
            float bv = bg[l*D_MODEL + nf*16 + l16];
            v8f acc;
            #pragma unroll
            for (int i = 0; i < 8; ++i) acc[i] = bv;
            #pragma unroll
            for (int kt = 0; kt < 4; ++kt) {
                v16bf bF = load_frag_b(Wgl, kt*8 + nf, lane);
                acc = __builtin_amdgcn_wmma_f32_16x16x32_bf16(
                          false, aF[kt], false, bF, (short)0, acc, false, false);
            }
            #pragma unroll
            for (int i = 0; i < 8; ++i) acc[i] = fmaxf(acc[i], 0.0f);
            c[nf] = acc;
        }

        // ---- LayerNorm on fragments (row i lives in half-wave; reduce lanes) ----
        float mu[8], rstd[8];
        #pragma unroll
        for (int i = 0; i < 8; ++i) {
            float s = 0.0f;
            #pragma unroll
            for (int nf = 0; nf < 8; ++nf) s += c[nf][i];
            s += __shfl_xor(s, 1, 32); s += __shfl_xor(s, 2, 32);
            s += __shfl_xor(s, 4, 32); s += __shfl_xor(s, 8, 32);
            mu[i] = s * (1.0f/128.0f);
            float v = 0.0f;
            #pragma unroll
            for (int nf = 0; nf < 8; ++nf) { float d = c[nf][i] - mu[i]; v += d*d; }
            v += __shfl_xor(v, 1, 32); v += __shfl_xor(v, 2, 32);
            v += __shfl_xor(v, 4, 32); v += __shfl_xor(v, 8, 32);
            rstd[i] = rsqrtf(v * (1.0f/128.0f) + LN_EPS);
        }
        #pragma unroll
        for (int nf = 0; nf < 8; ++nf) {
            int col = nf*16 + l16;
            float gv = ln_g[l*D_MODEL + col], bv = ln_b[l*D_MODEL + col];
            #pragma unroll
            for (int i = 0; i < 8; ++i)
                c[nf][i] = (c[nf][i] - mu[i]) * rstd[i] * gv + bv;
        }

        // store h_ln (own rows) and reload as A fragments
        #pragma unroll
        for (int nf = 0; nf < 8; ++nf) {
            int col = nf*16 + l16;
            #pragma unroll
            for (int i = 0; i < 8; ++i)
                hB[(row0 + (lhalf ? i+8 : i))*STR + col] = (__bf16)c[nf][i];
        }
        const __bf16* hRow = hB + (row0 + l16)*STR;
        v16bf hF[4];
        #pragma unroll
        for (int kt = 0; kt < 4; ++kt) hF[kt] = load_frag_a(hRow, kt, lhalf);

        // ---- residual FFN: h + relu(h@W1+b1)@W2 + b2, streamed in 128-col chunks
        __bf16* tB = (__bf16*)(smem + OFF_SCR) + wave*(16*STR);  // wave-private
        v8f c2[8];
        #pragma unroll
        for (int nf2 = 0; nf2 < 8; ++nf2) {
            float bv = b2[l*D_MODEL + nf2*16 + l16];
            #pragma unroll
            for (int i = 0; i < 8; ++i) c2[nf2][i] = c[nf2][i] + bv;
        }
        #pragma unroll 1
        for (int ch = 0; ch < 4; ++ch) {
            // t = relu(h_ln @ W1[:, ch*128:+128] + b1)
            #pragma unroll
            for (int nf1 = 0; nf1 < 8; ++nf1) {
                float bv = b1[l*HID + ch*128 + nf1*16 + l16];
                v8f a1;
                #pragma unroll
                for (int i = 0; i < 8; ++i) a1[i] = bv;
                #pragma unroll
                for (int kt = 0; kt < 4; ++kt) {
                    v16bf bF = load_frag_b(W1l, kt*32 + ch*8 + nf1, lane);
                    a1 = __builtin_amdgcn_wmma_f32_16x16x32_bf16(
                             false, hF[kt], false, bF, (short)0, a1, false, false);
                }
                int col = nf1*16 + l16;
                #pragma unroll
                for (int i = 0; i < 8; ++i)
                    tB[(lhalf ? i+8 : i)*STR + col] = (__bf16)fmaxf(a1[i], 0.0f);
            }
            // c2 += relu(t) @ W2[ch*128:+128, :]
            const __bf16* tRow = tB + l16*STR;
            v16bf tF[4];
            #pragma unroll
            for (int kt = 0; kt < 4; ++kt) tF[kt] = load_frag_a(tRow, kt, lhalf);
            #pragma unroll
            for (int nf2 = 0; nf2 < 8; ++nf2) {
                #pragma unroll
                for (int kt = 0; kt < 4; ++kt) {
                    v16bf bF = load_frag_b(W2l, (ch*4 + kt)*8 + nf2, lane);
                    c2[nf2] = __builtin_amdgcn_wmma_f32_16x16x32_bf16(
                                  false, tF[kt], false, bF, (short)0, c2[nf2], false, false);
                }
            }
        }
        // write h_next (own rows)
        #pragma unroll
        for (int nf2 = 0; nf2 < 8; ++nf2) {
            int col = nf2*16 + l16;
            #pragma unroll
            for (int i = 0; i < 8; ++i)
                hB[(row0 + (lhalf ? i+8 : i))*STR + col] = (__bf16)c2[nf2][i];
        }
        __syncthreads();   // scratch + h hazards across waves for next layer
    }

    // ---- readout: mean over 9 nodes, fc, sigmoid ----
    float* red = scr;                       // 16 graphs x 16 partials
    int g = tid >> 4, sub = tid & 15;
    if (g < GPW) {
        float acc = 0.0f;
        for (int cc = sub*8; cc < sub*8 + 8; ++cc) {
            float s = 0.0f;
            #pragma unroll
            for (int r = 0; r < NPG; ++r) s += (float)hB[(g*NPG + r)*STR + cc];
            acc += (s * (1.0f/9.0f)) * fc_w[cc];
        }
        red[g*16 + sub] = acc;
    }
    __syncthreads();
    if (tid < GPW) {
        float s = fc_b[0];
        #pragma unroll
        for (int k = 0; k < 16; ++k) s += red[tid*16 + k];
        out[g0 + tid] = 1.0f / (1.0f + __expf(-s));
    }
}

extern "C" void kernel_launch(void* const* d_in, const int* in_sizes, int n_in,
                              void* d_out, int out_size, void* d_ws, size_t ws_size,
                              hipStream_t stream) {
    const float* op_table   = (const float*)d_in[0];
    const float* device_emb = (const float*)d_in[1];
    const float* Wg         = (const float*)d_in[2];
    const float* bg         = (const float*)d_in[3];
    const float* ln_g       = (const float*)d_in[4];
    const float* ln_b       = (const float*)d_in[5];
    const float* W1         = (const float*)d_in[6];
    const float* b1         = (const float*)d_in[7];
    const float* W2         = (const float*)d_in[8];
    const float* b2         = (const float*)d_in[9];
    const float* fc_w       = (const float*)d_in[10];
    const float* fc_b       = (const float*)d_in[11];
    const int*   op_idx     = (const int*)d_in[12];
    const int*   srcE       = (const int*)d_in[13];
    const int*   dstE       = (const int*)d_in[14];

    __bf16* wsW = (__bf16*)d_ws;

    // pre-swizzle weights to bf16 WMMA-B fragment layout (runs every call; deterministic)
    prep_weights<<<dim3(64, NLAYERS), 256, 0, stream>>>(Wg, wsW + WG_OFF, D_MODEL, D_MODEL);
    prep_weights<<<dim3(256, NLAYERS), 256, 0, stream>>>(W1, wsW + W1_OFF, D_MODEL, HID);
    prep_weights<<<dim3(256, NLAYERS), 256, 0, stream>>>(W2, wsW + W2_OFF, HID, D_MODEL);

    const int B = in_sizes[12] / NPG;          // 16384
    gcn_fused_kernel<<<B / GPW, NTHREADS, SMEM_BYTES, stream>>>(
        op_table, device_emb, bg, ln_g, ln_b, b1, b2, fc_w, fc_b,
        op_idx, srcE, dstE, wsW, (float*)d_out);
}
